// MultiHeadAttention_17317308137927
// MI455X (gfx1250) — matrix operand order repaired and checked by reading, not compiled, not run on hardware
//
#include <hip/hip_runtime.h>
#include <hip/hip_bf16.h>

typedef __bf16 bf16;
typedef __attribute__((ext_vector_type(16))) __bf16 v16bf;
typedef __attribute__((ext_vector_type(8)))  __bf16 v8bf;
typedef __attribute__((ext_vector_type(8)))  float   v8f;

#define WMMA_BF16(a, b, c) \
  __builtin_amdgcn_wmma_f32_16x16x32_bf16(false, (a), false, (b), (short)0, (c), false, false)

// Interleave hint for one unrolled GEMM stage: 24 VMEM reads : 16 WMMAs
// expressed as 8 groups of (3 VMEM_READ, 2 MFMA/WMMA).
__device__ __forceinline__ void sched_stage() {
#if __has_builtin(__builtin_amdgcn_sched_group_barrier)
#pragma unroll
  for (int i = 0; i < 8; ++i) {
    __builtin_amdgcn_sched_group_barrier(0x020, 3, 0);  // VMEM read
    __builtin_amdgcn_sched_group_barrier(0x008, 2, 0);  // matrix (MFMA/WMMA)
  }
#endif
}

// ---------------------------------------------------------------------------
// Fragment loaders (CDNA5 wave32 WMMA layouts, ISA 7.12.2)
// ---------------------------------------------------------------------------

// A matrix 16x32 bf16, row-major source with leading dim `ld`.
// lane: m = lane&15, half = lane>>4.  Elements e=0..7 -> K = half*8 + e,
// e=8..15 -> K = 16 + half*8 + (e-8).  Two aligned 16B loads per lane.
__device__ __forceinline__ v16bf load_A_frag(const bf16* base, int ld, int row0, int k0) {
  const int lane = threadIdx.x & 31;
  const int m = lane & 15, half = lane >> 4;
  const bf16* p = base + (size_t)(row0 + m) * ld + k0 + half * 8;
  v8bf lo = *(const v8bf*)p;
  v8bf hi = *(const v8bf*)(p + 16);
  v16bf r;
#pragma unroll
  for (int i = 0; i < 8; ++i) { r[i] = lo[i]; r[8 + i] = hi[i]; }
  return r;
}

// B matrix 32x16 bf16 where B[kk][n] = W[n0+n][k0+kk], W row-major, ld elems.
// lane: n = lane&15, half = lane>>4; e=0..7 -> kk = half*8 + e, e=8..15 -> kk = 16 + half*8 + (e-8).
__device__ __forceinline__ v16bf load_B_frag(const bf16* W, int ld, int n0, int k0) {
  const int lane = threadIdx.x & 31;
  const int n = lane & 15, half = lane >> 4;
  const bf16* p = W + (size_t)(n0 + n) * ld + k0 + half * 8;
  v8bf lo = *(const v8bf*)p;
  v8bf hi = *(const v8bf*)(p + 16);
  v16bf r;
#pragma unroll
  for (int i = 0; i < 8; ++i) { r[i] = lo[i]; r[8 + i] = hi[i]; }
  return r;
}

// C/D 16x16 f32: element r lives at row (r + 8*half), col (lane&15).

// ---------------------------------------------------------------------------
// Precision-convert kernels
// ---------------------------------------------------------------------------
__global__ void cvt_f32_bf16(const float* __restrict__ src, bf16* __restrict__ dst, long n) {
  long stride = (long)gridDim.x * blockDim.x;
  for (long i = (long)blockIdx.x * blockDim.x + threadIdx.x; i < n; i += stride)
    dst[i] = (bf16)src[i];
}

// src (H, C, D) fp32 -> dst (H, D, C) bf16  (so B-fragments are row-contiguous)
__global__ void cvt_f32_bf16_T(const float* __restrict__ src, bf16* __restrict__ dst,
                               int H, int C, int D) {
  long n = (long)H * C * D;
  long stride = (long)gridDim.x * blockDim.x;
  for (long i = (long)blockIdx.x * blockDim.x + threadIdx.x; i < n; i += stride) {
    int d = (int)(i % D);
    long t = i / D;
    int c = (int)(t % C);
    int h = (int)(t / C);
    dst[((long)h * D + d) * C + c] = (bf16)src[i];
  }
}

// ---------------------------------------------------------------------------
// Kernel 1: QKV projection.  Per (mat,b,h): out(2048x64) = x_b(2048x1024) * W
// Wave tile 32x64 (2 A-frags x 4 B-frags -> 8 WMMAs per 32-K step; each B
// fragment feeds 2 WMMAs).  Fully-unrolled ping-pong K loop + sched hints.
// q,k stored (B,H,T,64); v stored transposed (B,H,64,T).
// ---------------------------------------------------------------------------
__global__ __launch_bounds__(256) void qkv_kernel(
    const bf16* __restrict__ xb,
    const bf16* __restrict__ wqT, const bf16* __restrict__ wkT, const bf16* __restrict__ wvT,
    bf16* __restrict__ q, bf16* __restrict__ k, bf16* __restrict__ vT) {
  const int T = 2048, C = 1024, D = 64, H = 16;
  int idx = blockIdx.x;
  const int qt = idx & 7;   idx >>= 3;   // 8 row-groups of 256
  const int h  = idx & 15;  idx >>= 4;   // head
  const int b  = idx & 3;   idx >>= 2;   // batch
  const int mat = idx;                   // 0=q 1=k 2=v
  const int wave = threadIdx.x >> 5;
  const int lane = threadIdx.x & 31, nl = lane & 15, half = lane >> 4;
  const int row0 = qt * 256 + wave * 32;

  const bf16* A = xb + (size_t)b * T * C;
  const bf16* W = (mat == 0 ? wqT : mat == 1 ? wkT : wvT) + (size_t)h * D * C;

  v8f acc[8] = {};                       // [m*4 + j]
  v16bf a0[2], b0[4];
  a0[0] = load_A_frag(A, C, row0, 0);
  a0[1] = load_A_frag(A, C, row0 + 16, 0);
#pragma unroll
  for (int j = 0; j < 4; ++j) b0[j] = load_B_frag(W, C, j * 16, 0);

#pragma unroll
  for (int k0 = 0; k0 < C; k0 += 64) {
    v16bf a1[2], b1[4];
    a1[0] = load_A_frag(A, C, row0, k0 + 32);
    a1[1] = load_A_frag(A, C, row0 + 16, k0 + 32);
#pragma unroll
    for (int j = 0; j < 4; ++j) b1[j] = load_B_frag(W, C, j * 16, k0 + 32);
#pragma unroll
    for (int j = 0; j < 4; ++j) {
      acc[j]     = WMMA_BF16(a0[0], b0[j], acc[j]);
      acc[4 + j] = WMMA_BF16(a0[1], b0[j], acc[4 + j]);
    }

    const int k2 = (k0 + 64 < C) ? k0 + 64 : k0;   // dead prefetch on last stage
    a0[0] = load_A_frag(A, C, row0, k2);
    a0[1] = load_A_frag(A, C, row0 + 16, k2);
#pragma unroll
    for (int j = 0; j < 4; ++j) b0[j] = load_B_frag(W, C, j * 16, k2);
#pragma unroll
    for (int j = 0; j < 4; ++j) {
      acc[j]     = WMMA_BF16(a1[0], b1[j], acc[j]);
      acc[4 + j] = WMMA_BF16(a1[1], b1[j], acc[4 + j]);
    }
    sched_stage();
  }

  if (mat < 2) {
    bf16* out = (mat == 0 ? q : k) + (size_t)(b * H + h) * T * D;
#pragma unroll
    for (int m = 0; m < 2; ++m)
#pragma unroll
      for (int j = 0; j < 4; ++j)
#pragma unroll
        for (int r = 0; r < 8; ++r)
          out[(size_t)(row0 + m * 16 + r + 8 * half) * D + j * 16 + nl] =
              (bf16)acc[m * 4 + j][r];
  } else {
    bf16* out = vT + (size_t)(b * H + h) * D * T;
#pragma unroll
    for (int m = 0; m < 2; ++m)
#pragma unroll
      for (int j = 0; j < 4; ++j)
#pragma unroll
        for (int r = 0; r < 8; ++r)
          out[(size_t)(j * 16 + nl) * T + (row0 + m * 16 + r + 8 * half)] =
              (bf16)acc[m * 4 + j][r];
  }
}

// ---------------------------------------------------------------------------
// Flash-attention block body: V frags issued first (consumed after softmax),
// score WMMAs on K frags loaded one block earlier, online softmax, P C-layout
// -> A-layout via wave-private LDS, then P*V WMMAs.
// ---------------------------------------------------------------------------
__device__ __forceinline__ void fa_block(
    int s0, int qBase, int nl, int half,
    const v16bf& qA0, const v16bf& qA1,
    const v16bf& kf0, const v16bf& kf1, const v16bf& kf2, const v16bf& kf3,
    const bf16* __restrict__ vp, bf16* myP,
    v8f (&yacc)[4], float (&mrow)[8], float (&lrow)[8]) {
  const float scale = 0.125f;            // 1/sqrt(64)

  // issue V fragments early: B[kk][j] = v[s0+kk][j] = vT[j][s0+kk]
  v16bf vf[4];
#pragma unroll
  for (int j = 0; j < 4; ++j) vf[j] = load_B_frag(vp, 2048, j * 16, s0);

  // scores: two 16x16 tiles = q(16x64) * k^T
  v8f sc0 = {}, sc1 = {};
  sc0 = WMMA_BF16(qA0, kf0, sc0);
  sc0 = WMMA_BF16(qA1, kf1, sc0);
  sc1 = WMMA_BF16(qA0, kf2, sc1);
  sc1 = WMMA_BF16(qA1, kf3, sc1);

  // online softmax per row (row = r + 8*half spans 16 lanes of one half)
#pragma unroll
  for (int r = 0; r < 8; ++r) {
    const int qr = qBase + r + 8 * half;
    float a0 = sc0[r] * scale, a1 = sc1[r] * scale;
    const bool ok0 = (s0 + nl) <= qr, ok1 = (s0 + 16 + nl) <= qr;
    a0 = ok0 ? a0 : -1e30f;
    a1 = ok1 ? a1 : -1e30f;
    float mx = fmaxf(a0, a1);
#pragma unroll
    for (int d = 1; d < 16; d <<= 1) mx = fmaxf(mx, __shfl_xor(mx, d, 32));
    const float newm = fmaxf(mrow[r], mx);
    const float alpha = __expf(mrow[r] - newm);
    const float p0 = ok0 ? __expf(a0 - newm) : 0.f;
    const float p1 = ok1 ? __expf(a1 - newm) : 0.f;
    float ps = p0 + p1;
#pragma unroll
    for (int d = 1; d < 16; d <<= 1) ps += __shfl_xor(ps, d, 32);
    lrow[r] = lrow[r] * alpha + ps;
    mrow[r] = newm;
#pragma unroll
    for (int j = 0; j < 4; ++j) yacc[j][r] *= alpha;
    const int row = r + 8 * half;
    myP[row * 32 + nl]      = (bf16)p0;
    myP[row * 32 + 16 + nl] = (bf16)p1;
  }
  // wave-private LDS region: our own DS stores are in-order within the wave
  asm volatile("s_wait_dscnt 0x0" ::: "memory");
  const v16bf pA = load_A_frag(myP, 32, 0, 0);
#pragma unroll
  for (int j = 0; j < 4; ++j)
    yacc[j] = WMMA_BF16(pA, vf[j], yacc[j]);
}

// ---------------------------------------------------------------------------
// Kernel 2: causal flash attention per (b,h).  Wave = 16 query rows, key
// blocks of 32.  2x-unrolled ping-pong: K-set A / K-set B alternate so the
// prefetch loads write the consuming registers of the *next* block directly
// (no rotation movs, no WAR hazard nops).  EXEC stays full throughout.
// ---------------------------------------------------------------------------
__global__ __launch_bounds__(256) void flash_kernel(
    const bf16* __restrict__ q, const bf16* __restrict__ k,
    const bf16* __restrict__ vT, bf16* __restrict__ yb) {
  const int T = 2048, D = 64, H = 16, C = 1024;
  __shared__ bf16 ldsP[8][16 * 32];

  const int qt = blockIdx.x & 15;        // 16 query groups of 128 rows
  const int bh = blockIdx.x >> 4;        // (b*H + h), 0..63
  const int b = bh >> 4, h = bh & 15;
  const int wave = threadIdx.x >> 5;
  const int lane = threadIdx.x & 31, nl = lane & 15, half = lane >> 4;
  const int qBase = qt * 128 + wave * 16;

  const bf16* qp = q  + (size_t)bh * T * D;
  const bf16* kp = k  + (size_t)bh * T * D;
  const bf16* vp = vT + (size_t)bh * D * T;

  const v16bf qA0 = load_A_frag(qp, D, qBase, 0);
  const v16bf qA1 = load_A_frag(qp, D, qBase, 32);

  v8f yacc[4] = {};
  float mrow[8], lrow[8];
#pragma unroll
  for (int r = 0; r < 8; ++r) { mrow[r] = -1e30f; lrow[r] = 0.f; }

  bf16* myP = &ldsP[wave][0];
  const int maxS0 = ((qBase + 15) >> 5) << 5;   // last causal key-block start

  // ping set A: K fragments for block 0
  v16bf kA0 = load_B_frag(kp, D, 0, 0);
  v16bf kA1 = load_B_frag(kp, D, 0, 32);
  v16bf kA2 = load_B_frag(kp, D, 16, 0);
  v16bf kA3 = load_B_frag(kp, D, 16, 32);

  int s0 = 0;
  while (s0 + 32 <= maxS0) {             // uniform per wave
    // prefetch K for block s0+32 into pong set B
    v16bf kB0 = load_B_frag(kp, D, s0 + 32, 0);
    v16bf kB1 = load_B_frag(kp, D, s0 + 32, 32);
    v16bf kB2 = load_B_frag(kp, D, s0 + 48, 0);
    v16bf kB3 = load_B_frag(kp, D, s0 + 48, 32);
    fa_block(s0, qBase, nl, half, qA0, qA1, kA0, kA1, kA2, kA3, vp, myP,
             yacc, mrow, lrow);

    // prefetch K for block s0+64 (clamped) back into ping set A
    const int s2 = (s0 + 64 <= maxS0) ? s0 + 64 : maxS0;
    kA0 = load_B_frag(kp, D, s2, 0);
    kA1 = load_B_frag(kp, D, s2, 32);
    kA2 = load_B_frag(kp, D, s2 + 16, 0);
    kA3 = load_B_frag(kp, D, s2 + 16, 32);
    fa_block(s0 + 32, qBase, nl, half, qA0, qA1, kB0, kB1, kB2, kB3, vp, myP,
             yacc, mrow, lrow);
    s0 += 64;
  }
  if (s0 <= maxS0)                        // final (odd) block: set A holds it
    fa_block(s0, qBase, nl, half, qA0, qA1, kA0, kA1, kA2, kA3, vp, myP,
             yacc, mrow, lrow);

  // epilogue: normalize and write concat-head layout (B,T,C) at col h*64
  bf16* yo = yb + (size_t)b * T * C + (size_t)h * D;
#pragma unroll
  for (int r = 0; r < 8; ++r) {
    const float inv = 1.f / lrow[r];
    const int row = qBase + r + 8 * half;
#pragma unroll
    for (int j = 0; j < 4; ++j)
      yo[(size_t)row * C + j * 16 + nl] = (bf16)(yacc[j][r] * inv);
  }
}

// ---------------------------------------------------------------------------
// Kernel 3: output projection  out(8192x1024) = y(8192x1024) @ w_proj^T + b
// Wave tile 32x64, fully-unrolled pipelined K loop + sched hints.  fp32 out.
// ---------------------------------------------------------------------------
__global__ __launch_bounds__(256) void proj_kernel(
    const bf16* __restrict__ yb, const bf16* __restrict__ wpb,
    const float* __restrict__ bias, float* __restrict__ out) {
  const int C = 1024;
  const int nblk = blockIdx.x & 15;      // 16 col groups of 64
  const int mblk = blockIdx.x >> 4;      // 32 row groups of 256
  const int wave = threadIdx.x >> 5;
  const int lane = threadIdx.x & 31, nl = lane & 15, half = lane >> 4;
  const int row0 = mblk * 256 + wave * 32;
  const int col0 = nblk * 64;

  v8f acc[8] = {};
  v16bf a0[2], b0[4];
  a0[0] = load_A_frag(yb, C, row0, 0);
  a0[1] = load_A_frag(yb, C, row0 + 16, 0);
#pragma unroll
  for (int j = 0; j < 4; ++j) b0[j] = load_B_frag(wpb, C, col0 + j * 16, 0);

#pragma unroll
  for (int k0 = 0; k0 < C; k0 += 64) {
    v16bf a1[2], b1[4];
    a1[0] = load_A_frag(yb, C, row0, k0 + 32);
    a1[1] = load_A_frag(yb, C, row0 + 16, k0 + 32);
#pragma unroll
    for (int j = 0; j < 4; ++j) b1[j] = load_B_frag(wpb, C, col0 + j * 16, k0 + 32);
#pragma unroll
    for (int j = 0; j < 4; ++j) {
      acc[j]     = WMMA_BF16(a0[0], b0[j], acc[j]);
      acc[4 + j] = WMMA_BF16(a0[1], b0[j], acc[4 + j]);
    }

    const int k2 = (k0 + 64 < C) ? k0 + 64 : k0;
    a0[0] = load_A_frag(yb, C, row0, k2);
    a0[1] = load_A_frag(yb, C, row0 + 16, k2);
#pragma unroll
    for (int j = 0; j < 4; ++j) b0[j] = load_B_frag(wpb, C, col0 + j * 16, k2);
#pragma unroll
    for (int j = 0; j < 4; ++j) {
      acc[j]     = WMMA_BF16(a1[0], b1[j], acc[j]);
      acc[4 + j] = WMMA_BF16(a1[1], b1[j], acc[4 + j]);
    }
    sched_stage();
  }

#pragma unroll
  for (int j = 0; j < 4; ++j) {
    const float bv = bias[col0 + j * 16 + nl];
#pragma unroll
    for (int m = 0; m < 2; ++m)
#pragma unroll
      for (int r = 0; r < 8; ++r)
        out[(size_t)(row0 + m * 16 + r + 8 * half) * C + col0 + j * 16 + nl] =
            acc[m * 4 + j][r] + bv;
  }
}

// ---------------------------------------------------------------------------
// Host launcher
// ---------------------------------------------------------------------------
extern "C" void kernel_launch(void* const* d_in, const int* in_sizes, int n_in,
                              void* d_out, int out_size, void* d_ws, size_t ws_size,
                              hipStream_t stream) {
  (void)in_sizes; (void)n_in; (void)out_size; (void)ws_size;
  const int B = 4, T = 2048, C = 1024, H = 16, D = 64;

  const float* x      = (const float*)d_in[0];
  const float* wq     = (const float*)d_in[1];
  const float* wk     = (const float*)d_in[2];
  const float* wv     = (const float*)d_in[3];
  const float* w_proj = (const float*)d_in[4];
  const float* b_proj = (const float*)d_in[5];
  float* out = (float*)d_out;

  // workspace layout (bf16 elements)
  bf16* ws = (bf16*)d_ws;
  const size_t nx  = (size_t)B * T * C;      // 8,388,608
  const size_t nw  = (size_t)H * C * D;      // 1,048,576
  const size_t nwp = (size_t)C * C;          // 1,048,576
  bf16* xb  = ws;                 size_t off = nx;
  bf16* wqT = ws + off;           off += nw;
  bf16* wkT = ws + off;           off += nw;
  bf16* wvT = ws + off;           off += nw;
  bf16* wpb = ws + off;           off += nwp;
  bf16* qb  = ws + off;           off += nx;  // (B,H,T,D)
  bf16* kb  = ws + off;           off += nx;  // (B,H,T,D)
  bf16* vTb = ws + off;           off += nx;  // (B,H,64,T)
  bf16* yb  = ws + off;           off += nx;  // (B,T,C)

  // 1) precision conversion (+ weight transposes for contiguous B-fragments)
  cvt_f32_bf16   <<<2048, 256, 0, stream>>>(x, xb, (long)nx);
  cvt_f32_bf16_T <<< 512, 256, 0, stream>>>(wq, wqT, H, C, D);
  cvt_f32_bf16_T <<< 512, 256, 0, stream>>>(wk, wkT, H, C, D);
  cvt_f32_bf16_T <<< 512, 256, 0, stream>>>(wv, wvT, H, C, D);
  cvt_f32_bf16   <<< 512, 256, 0, stream>>>(w_proj, wpb, (long)nwp);

  // 2) QKV projections: 3 mats * B*H * (T/256) blocks
  qkv_kernel<<<3 * B * H * (T / 256), 256, 0, stream>>>(xb, wqT, wkT, wvT, qb, kb, vTb);

  // 3) causal flash attention: B*H * (T/128) blocks
  flash_kernel<<<B * H * (T / 128), 256, 0, stream>>>(qb, kb, vTb, yb);

  // 4) output projection (+bias), fp32 result
  proj_kernel<<<(B * T / 256) * (C / 64), 256, 0, stream>>>(yb, wpb, b_proj, out);
}
